// MultiHeadAttention_24867860643991
// MI455X (gfx1250) — matrix operand order repaired and checked by reading, not compiled, not run on hardware
//
#include <hip/hip_runtime.h>

// ---------------------------------------------------------------------------
// Fused MHA for MI455X (gfx1250, wave32, WMMA).
//   B=2, T=2048, E=2048, H=16, D=128
// Pipeline:
//   1) cvt: x -> bf16 (row major), W_qkv/W_out -> bf16 transposed (N x K)
//   2) gemm_qkv: X(4096x2048) @ Wqkv^T -> Q (B,H,T,D), K (B,H,T,D), V^T (B,H,D,T), bf16
//   3) attn: flash-style causal attention per 16-row q tile -> ctx (B,T,E) bf16
//   4) gemm_out: ctx @ Wout^T + b_out -> out (B,T,E) fp32
// GEMMs: 32x64 tile per wave, register double-buffered (loads for k+32 issued
// before WMMAs for k), so s_wait_loadcnt covers a full iteration of latency.
// Workspace: ~113 MB of d_ws; hot working set fits in the 192 MB L2.
// ---------------------------------------------------------------------------

typedef __attribute__((ext_vector_type(16))) __bf16 v16bf;
typedef __attribute__((ext_vector_type(8)))  float  v8f;

#define BB 2
#define TT 2048
#define EE 2048
#define HH 16
#define DD 128
#define E3 (3 * EE)

union FragU { v16bf v; uint4 q[2]; };
union Pack8 { __bf16 h[8]; uint4 q; };

__device__ inline v8f vzero8() {
  v8f z;
#pragma unroll
  for (int i = 0; i < 8; ++i) z[i] = 0.0f;
  return z;
}

__device__ inline v8f wmma_bf16(v16bf a, v16bf b, v8f c) {
  return __builtin_amdgcn_wmma_f32_16x16x32_bf16(false, a, false, b, (short)0, c,
                                                 false, false);
}

// A fragment (16x32, bf16): lane<16 holds row (lane&15), K {0..7,16..23};
// lane>=16 holds same row, K {8..15,24..31}. base = first row of tile, ld in elems.
__device__ inline v16bf load_a(const __bf16* __restrict__ base, int ld, int k0, int lane) {
  const int half = lane >> 4;
  const int m    = lane & 15;
  const __bf16* p = base + (size_t)m * ld + k0 + half * 8;
  FragU f;
  f.q[0] = *(const uint4*)p;
  f.q[1] = *(const uint4*)(p + 16);
  return f.v;
}

// B fragment (32x16, bf16) from an N-major (N x K) buffer: lane holds column
// n = lane&15, K {0..7,16..23} (lane<16) or K {8..15,24..31} (lane>=16).
__device__ inline v16bf load_bt(const __bf16* __restrict__ base, int ld, int k0, int lane) {
  const int half = lane >> 4;
  const int n    = lane & 15;
  const __bf16* p = base + (size_t)n * ld + k0 + half * 8;
  FragU f;
  f.q[0] = *(const uint4*)p;
  f.q[1] = *(const uint4*)(p + 16);
  return f.v;
}

__device__ inline float rmax16(float v) {
#pragma unroll
  for (int off = 1; off < 16; off <<= 1) v = fmaxf(v, __shfl_xor(v, off, 32));
  return v;
}
__device__ inline float rsum16(float v) {
#pragma unroll
  for (int off = 1; off < 16; off <<= 1) v += __shfl_xor(v, off, 32);
  return v;
}

// ---------------------------------------------------------------------------
// Conversion kernels
// ---------------------------------------------------------------------------
// n must be a multiple of 8; each thread converts 8 floats -> 8 bf16 (one b128 store).
__global__ void cvt_copy8(const float* __restrict__ src, __bf16* __restrict__ dst, int n8) {
  int i = blockIdx.x * blockDim.x + threadIdx.x;
  if (i < n8) {
    const float4 lo = ((const float4*)src)[2 * i];
    const float4 hi = ((const float4*)src)[2 * i + 1];
    Pack8 p;
    p.h[0] = (__bf16)lo.x; p.h[1] = (__bf16)lo.y;
    p.h[2] = (__bf16)lo.z; p.h[3] = (__bf16)lo.w;
    p.h[4] = (__bf16)hi.x; p.h[5] = (__bf16)hi.y;
    p.h[6] = (__bf16)hi.z; p.h[7] = (__bf16)hi.w;
    ((uint4*)dst)[i] = p.q;
  }
}

// src is K x N row-major fp32; dst is N x K bf16 (dst[n*K + k] = src[k*N + n])
__global__ void cvt_transpose(const float* __restrict__ src, __bf16* __restrict__ dst,
                              int K, int N) {
  int i = blockIdx.x * blockDim.x + threadIdx.x;
  if (i < K * N) {
    int n = i / K;
    int k = i - n * K;
    dst[i] = (__bf16)src[(size_t)k * N + n];
  }
}

// ---------------------------------------------------------------------------
// GEMM main loop: wave computes 32 (M) x 64 (N); register double-buffered.
// acc0 = rows [0,16), acc1 = rows [16,32).
// ---------------------------------------------------------------------------
__device__ inline void mma_loop_32x64(const __bf16* __restrict__ arow,
                                      const __bf16* __restrict__ btrow,
                                      int kdim, v8f acc0[4], v8f acc1[4], int lane) {
  v16bf a0 = load_a(arow, kdim, 0, lane);
  v16bf a1 = load_a(arow + (size_t)16 * kdim, kdim, 0, lane);
  v16bf b0 = load_bt(btrow, kdim, 0, lane);
  v16bf b1 = load_bt(btrow + (size_t)16 * kdim, kdim, 0, lane);
  v16bf b2 = load_bt(btrow + (size_t)32 * kdim, kdim, 0, lane);
  v16bf b3 = load_bt(btrow + (size_t)48 * kdim, kdim, 0, lane);
  for (int k0 = 32; k0 < kdim; k0 += 32) {
    // Prefetch next K-step into registers before consuming current step.
    v16bf na0 = load_a(arow, kdim, k0, lane);
    v16bf na1 = load_a(arow + (size_t)16 * kdim, kdim, k0, lane);
    v16bf nb0 = load_bt(btrow, kdim, k0, lane);
    v16bf nb1 = load_bt(btrow + (size_t)16 * kdim, kdim, k0, lane);
    v16bf nb2 = load_bt(btrow + (size_t)32 * kdim, kdim, k0, lane);
    v16bf nb3 = load_bt(btrow + (size_t)48 * kdim, kdim, k0, lane);
    acc0[0] = wmma_bf16(a0, b0, acc0[0]);
    acc0[1] = wmma_bf16(a0, b1, acc0[1]);
    acc0[2] = wmma_bf16(a0, b2, acc0[2]);
    acc0[3] = wmma_bf16(a0, b3, acc0[3]);
    acc1[0] = wmma_bf16(a1, b0, acc1[0]);
    acc1[1] = wmma_bf16(a1, b1, acc1[1]);
    acc1[2] = wmma_bf16(a1, b2, acc1[2]);
    acc1[3] = wmma_bf16(a1, b3, acc1[3]);
    a0 = na0; a1 = na1; b0 = nb0; b1 = nb1; b2 = nb2; b3 = nb3;
  }
  acc0[0] = wmma_bf16(a0, b0, acc0[0]);
  acc0[1] = wmma_bf16(a0, b1, acc0[1]);
  acc0[2] = wmma_bf16(a0, b2, acc0[2]);
  acc0[3] = wmma_bf16(a0, b3, acc0[3]);
  acc1[0] = wmma_bf16(a1, b0, acc1[0]);
  acc1[1] = wmma_bf16(a1, b1, acc1[1]);
  acc1[2] = wmma_bf16(a1, b2, acc1[2]);
  acc1[3] = wmma_bf16(a1, b3, acc1[3]);
}

// ---------------------------------------------------------------------------
// GEMM1 epilogue for one 16x16 C fragment: scatter into Q/K/V(T).
// m0row = first token row of this 16-row fragment group.
// ---------------------------------------------------------------------------
__device__ inline void store_qkv_frag(const v8f& a, int c0, int m0row,
                                      const float* __restrict__ bias,
                                      __bf16* __restrict__ qo, __bf16* __restrict__ ko,
                                      __bf16* __restrict__ vto, int hl, int ln) {
  const int seg = c0 / EE;           // 0:Q 1:K 2:V
  const int rem = c0 - seg * EE;
  const int h   = rem >> 7;
  const int d   = (rem & 127) + ln;
  const float bv = bias[c0 + ln];
  const int btbase = m0row + 8 * hl;         // 8 consecutive token rows
  const int b = btbase >> 11;
  const int tbase = btbase & (TT - 1);
  const size_t head = (size_t)b * HH + h;
  if (seg == 2) {
    // V stored transposed (B,H,D,T): 8 consecutive t -> one b128 store.
    Pack8 p;
#pragma unroll
    for (int r = 0; r < 8; ++r) p.h[r] = (__bf16)(a[r] + bv);
    *(uint4*)(vto + (head * DD + d) * TT + tbase) = p.q;
  } else {
    __bf16* dst = (seg == 0) ? qo : ko;
#pragma unroll
    for (int r = 0; r < 8; ++r)
      dst[(head * TT + tbase + r) * DD + d] = (__bf16)(a[r] + bv);
  }
}

// ---------------------------------------------------------------------------
// GEMM1: block = 8 waves (2 M x 4 N), tile 64 (M) x 256 (N).
// Grid: (4096/64) * (6144/256) = 64 * 24 = 1536.
// ---------------------------------------------------------------------------
__global__ __launch_bounds__(256) void gemm_qkv(const __bf16* __restrict__ xb,
                                                const __bf16* __restrict__ wt,
                                                const float* __restrict__ bias,
                                                __bf16* __restrict__ qo,
                                                __bf16* __restrict__ ko,
                                                __bf16* __restrict__ vto) {
  const int lane = threadIdx.x & 31;
  const int wave = threadIdx.x >> 5;
  const int hl = lane >> 4, ln = lane & 15;
  const int tiles_n = E3 / 256;  // 24
  const int bm = blockIdx.x / tiles_n;
  const int bn = blockIdx.x - bm * tiles_n;
  const int m0 = bm * 64 + (wave >> 2) * 32;
  const int n0 = bn * 256 + (wave & 3) * 64;

  v8f acc0[4], acc1[4];
#pragma unroll
  for (int f = 0; f < 4; ++f) { acc0[f] = vzero8(); acc1[f] = vzero8(); }

  mma_loop_32x64(xb + (size_t)m0 * EE, wt + (size_t)n0 * EE, EE, acc0, acc1, lane);

#pragma unroll
  for (int f = 0; f < 4; ++f) {
    store_qkv_frag(acc0[f], n0 + 16 * f, m0,      bias, qo, ko, vto, hl, ln);
    store_qkv_frag(acc1[f], n0 + 16 * f, m0 + 16, bias, qo, ko, vto, hl, ln);
  }
}

// ---------------------------------------------------------------------------
// Flash attention: one wave per 16-row q tile; block = 4 waves (64 q rows).
// Online softmax in log2 domain; P C-layout -> A-layout via per-wave LDS bounce.
// ---------------------------------------------------------------------------
#define PST 40  // padded LDS row stride in halves (32 cols + 8 pad)

__global__ __launch_bounds__(128) void attn_kernel(const __bf16* __restrict__ qg,
                                                   const __bf16* __restrict__ kg,
                                                   const __bf16* __restrict__ vtg,
                                                   __bf16* __restrict__ ctx) {
  __shared__ __align__(16) __bf16 pt[4][16 * PST];
  const int lane = threadIdx.x & 31;
  const int wave = threadIdx.x >> 5;
  const int hl = lane >> 4, ln = lane & 15;
  const int nq = TT / 64;
  const int bh = blockIdx.x / nq;
  const int t0 = (blockIdx.x - bh * nq) * 64 + wave * 16;
  const int b = bh / HH, h = bh - b * HH;

  const __bf16* qp = qg  + (size_t)bh * TT * DD;
  const __bf16* kp = kg  + (size_t)bh * TT * DD;
  const __bf16* vp = vtg + (size_t)bh * DD * TT;
  __bf16* smem = &pt[wave][0];

  // Q fragments for all 4 K-chunks of D=128, loaded once.
  v16bf qa[4];
#pragma unroll
  for (int c = 0; c < 4; ++c) qa[c] = load_a(qp + (size_t)t0 * DD, DD, 32 * c, lane);

  v8f o[8];
#pragma unroll
  for (int f = 0; f < 8; ++f) o[f] = vzero8();
  float mrow[8], lrow[8];
#pragma unroll
  for (int r = 0; r < 8; ++r) { mrow[r] = -1e30f; lrow[r] = 0.0f; }

  // 1/sqrt(128) * log2(e): softmax computed with exp2.
  const float sscale = 0.088388347648318447f * 1.4426950408889634f;

  const int jmax = (t0 + 15) >> 5;  // inclusive, 32-wide key blocks
  for (int j32 = 0; j32 <= jmax; ++j32) {
    const int j = j32 << 5;
    if (j32 < jmax) {
      // Speculative L2 prefetch of the next key/value blocks (global_prefetch_b8).
      __builtin_prefetch(kp + (size_t)(j + 32) * DD, 0, 0);
      __builtin_prefetch(vp + j + 32, 0, 0);
    }
    // S = Q K^T for 16 x 32 keys (two 16x16 tiles), contraction over D=128.
    v8f s0 = vzero8(), s1 = vzero8();
#pragma unroll
    for (int c = 0; c < 4; ++c) {
      v16bf kb0 = load_bt(kp + (size_t)j * DD, DD, 32 * c, lane);
      s0 = wmma_bf16(qa[c], kb0, s0);
      v16bf kb1 = load_bt(kp + (size_t)(j + 16) * DD, DD, 32 * c, lane);
      s1 = wmma_bf16(qa[c], kb1, s1);
    }

    const bool need_mask = (j + 31 > t0);
    float alpha[8];
#pragma unroll
    for (int r = 0; r < 8; ++r) {
      float a0 = s0[r] * sscale;
      float a1 = s1[r] * sscale;
      if (need_mask) {
        const int trow = t0 + r + 8 * hl;
        if (j + ln > trow)      a0 = -1e30f;
        if (j + 16 + ln > trow) a1 = -1e30f;
      }
      float mx   = rmax16(fmaxf(a0, a1));
      float mnew = fmaxf(mrow[r], mx);
      alpha[r]   = exp2f(mrow[r] - mnew);
      a0 = exp2f(a0 - mnew);
      a1 = exp2f(a1 - mnew);
      lrow[r] = lrow[r] * alpha[r] + rsum16(a0 + a1);
      mrow[r] = mnew;
      s0[r] = a0;
      s1[r] = a1;
    }

    // Rescale running O by alpha (row-indexed, matches C layout).
#pragma unroll
    for (int f = 0; f < 8; ++f)
#pragma unroll
      for (int r = 0; r < 8; ++r) o[f][r] *= alpha[r];

    // P (C layout) -> LDS -> A fragment (16x32).
#pragma unroll
    for (int r = 0; r < 8; ++r) {
      const int row = r + 8 * hl;
      smem[row * PST + ln]      = (__bf16)s0[r];
      smem[row * PST + 16 + ln] = (__bf16)s1[r];
    }
    asm volatile("s_wait_dscnt 0" ::: "memory");
    v16bf pa;
    {
      const __bf16* pr = smem + ln * PST + hl * 8;
      FragU fr;
      fr.q[0] = *(const uint4*)pr;
      fr.q[1] = *(const uint4*)(pr + 16);
      pa = fr.v;
    }

    // O += P @ V, V^T stored (D,T) so B fragments are contiguous.
#pragma unroll
    for (int f = 0; f < 8; ++f) {
      v16bf vb = load_bt(vp + (size_t)(16 * f) * TT + j, TT, 0, lane);
      o[f] = wmma_bf16(pa, vb, o[f]);
    }
  }

  // Normalize and store ctx (B,T,E) bf16.
  float inv[8];
#pragma unroll
  for (int r = 0; r < 8; ++r) inv[r] = 1.0f / lrow[r];
  __bf16* cbase = ctx + (size_t)b * TT * EE + (size_t)h * DD;
#pragma unroll
  for (int f = 0; f < 8; ++f)
#pragma unroll
    for (int r = 0; r < 8; ++r) {
      const int t = t0 + r + 8 * hl;
      cbase[(size_t)t * EE + 16 * f + ln] = (__bf16)(o[f][r] * inv[r]);
    }
}

// ---------------------------------------------------------------------------
// GEMM2: out = ctx @ W_out + b_out, fp32 output.
// Block = 8 waves (2 M x 4 N), tile 64 x 256. Grid: 64 * 8 = 512.
// ---------------------------------------------------------------------------
__global__ __launch_bounds__(256) void gemm_out(const __bf16* __restrict__ cb,
                                                const __bf16* __restrict__ wt,
                                                const float* __restrict__ bias,
                                                float* __restrict__ out) {
  const int lane = threadIdx.x & 31;
  const int wave = threadIdx.x >> 5;
  const int hl = lane >> 4, ln = lane & 15;
  const int tiles_n = EE / 256;  // 8
  const int bm = blockIdx.x / tiles_n;
  const int bn = blockIdx.x - bm * tiles_n;
  const int m0 = bm * 64 + (wave >> 2) * 32;
  const int n0 = bn * 256 + (wave & 3) * 64;

  v8f acc0[4], acc1[4];
#pragma unroll
  for (int f = 0; f < 4; ++f) { acc0[f] = vzero8(); acc1[f] = vzero8(); }

  mma_loop_32x64(cb + (size_t)m0 * EE, wt + (size_t)n0 * EE, EE, acc0, acc1, lane);

#pragma unroll
  for (int f = 0; f < 4; ++f) {
    const int c0 = n0 + 16 * f;
    const float bv = bias[c0 + ln];
#pragma unroll
    for (int r = 0; r < 8; ++r) {
      const int bt0 = m0 + r + 8 * hl;
      const int bt1 = m0 + 16 + r + 8 * hl;
      out[(size_t)bt0 * EE + c0 + ln] = acc0[f][r] + bv;
      out[(size_t)bt1 * EE + c0 + ln] = acc1[f][r] + bv;
    }
  }
}

// ---------------------------------------------------------------------------
// Host launcher
// ---------------------------------------------------------------------------
extern "C" void kernel_launch(void* const* d_in, const int* in_sizes, int n_in,
                              void* d_out, int out_size, void* d_ws, size_t ws_size,
                              hipStream_t stream) {
  (void)in_sizes; (void)n_in; (void)out_size; (void)ws_size;
  const float* x    = (const float*)d_in[0];
  const float* Wqkv = (const float*)d_in[1];
  const float* bqkv = (const float*)d_in[2];
  const float* Wout = (const float*)d_in[3];
  const float* bout = (const float*)d_in[4];
  float* out = (float*)d_out;

  const size_t nx  = (size_t)BB * TT * EE;  // 8M elems
  const size_t nwq = (size_t)EE * E3;       // 12.58M
  const size_t nwo = (size_t)EE * EE;       // 4M

  __bf16* xb  = (__bf16*)d_ws;
  __bf16* wqt = xb + nx;
  __bf16* wot = wqt + nwq;
  __bf16* qb  = wot + nwo;
  __bf16* kb  = qb + nx;
  __bf16* vt  = kb + nx;
  __bf16* cx  = vt + nx;   // total ~113 MB of workspace

  cvt_copy8<<<(unsigned)((nx / 8 + 255) / 256), 256, 0, stream>>>(x, xb, (int)(nx / 8));
  cvt_transpose<<<(unsigned)((nwq + 255) / 256), 256, 0, stream>>>(Wqkv, wqt, EE, E3);
  cvt_transpose<<<(unsigned)((nwo + 255) / 256), 256, 0, stream>>>(Wout, wot, EE, EE);

  gemm_qkv<<<(BB * TT / 64) * (E3 / 256), 256, 0, stream>>>(xb, wqt, bqkv, qb, kb, vt);
  attn_kernel<<<BB * HH * (TT / 64), 128, 0, stream>>>(qb, kb, vt, cx);
  gemm_out<<<(BB * TT / 64) * (EE / 256), 256, 0, stream>>>(cx, wot, bout, out);
}
